// EdgeSetAttention_20246475833772
// MI455X (gfx1250) — matrix operand order repaired and checked by reading, not compiled, not run on hardware
//
#include <hip/hip_runtime.h>

// EdgeSetAttention for MI455X (gfx1250, wave32, WMMA f16 -> f32 accumulate).
// All heavy math through v_wmma_f32_16x16x32_f16. Flash-attention streaming so
// the 4x8192x8192 score tensor is never materialized (K/V live in L2).
// K/A/B tile staging uses GLOBAL_LOAD_ASYNC_TO_LDS_B128 when the builtin is
// available (ASYNCcnt + s_wait_asynccnt), else falls back to b128 ld/st.

typedef __attribute__((ext_vector_type(16))) _Float16 v16h;
typedef __attribute__((ext_vector_type(8)))  _Float16 v8h;
typedef __attribute__((ext_vector_type(4)))  _Float16 v4h;
typedef __attribute__((ext_vector_type(8)))  float    v8f;
typedef __attribute__((ext_vector_type(4)))  unsigned v4u;
typedef __attribute__((ext_vector_type(4)))  int      v4i;

#if __has_builtin(__builtin_amdgcn_global_load_async_to_lds_b128) && \
    __has_builtin(__builtin_amdgcn_s_wait_asynccnt)
#define HAS_ASYNC_LDS 1
#else
#define HAS_ASYNC_LDS 0
#endif

namespace {

constexpr int kH     = 256;
constexpr int kNH    = 4;
constexpr int kHD    = 64;
constexpr int kNodes = 2048;
constexpr int kEdges = 8192;

union FragH {
  v16h v;
  _Float16 h[16];
  unsigned u[8];
};

// 16B global -> LDS copy; async (ASYNCcnt) on gfx1250 toolchains that expose it.
// Builtin signature (from clang diagnostic): (v4i AS1*, v4i AS3*, imm off, imm cpol)
__device__ __forceinline__ void stage16(const _Float16* __restrict__ g,
                                        _Float16* l) {
#if HAS_ASYNC_LDS
  __builtin_amdgcn_global_load_async_to_lds_b128(
      (__attribute__((address_space(1))) v4i*)g,
      (__attribute__((address_space(3))) v4i*)l, 0, 0);
#else
  *(v8h*)l = *(const v8h*)g;
#endif
}
__device__ __forceinline__ void stage_wait() {
#if HAS_ASYNC_LDS
  __builtin_amdgcn_s_wait_asynccnt(0);
#endif
}

__device__ __forceinline__ v8f wmma_f32_f16(v16h a, v16h b, v8f c) {
  // 8 args: (neg_a, A, neg_b, B, c_mod, C, reuse_a, reuse_b)
  return __builtin_amdgcn_wmma_f32_16x16x32_f16(false, a, false, b, (short)0, c,
                                                false, false);
}

// A-matrix 16x32 fragment from LDS tile stored row-major [16][ld] (f16, k0 even
// -> dword loads). ISA 7.12.2: lane(0-15)=M, pairs k0=(i>=4?16:0)+half*8+2(i&3).
__device__ __forceinline__ v16h load_a_frag(const _Float16* base, int ld) {
  int lane = threadIdx.x & 31;
  int half = lane >> 4, m = lane & 15;
  FragH f;
#pragma unroll
  for (int i = 0; i < 8; ++i) {
    int k0 = ((i >> 2) << 4) + (half << 3) + ((i & 3) << 1);
    f.u[i] = *(const unsigned*)(base + m * ld + k0);
  }
  return f.v;
}

// B-matrix 32x16 fragment; tile stored [n][k] row-major (stride ld).
// Same k order as A: pairs k0=(i>=4?16:0)+half*8+2(i&3).
__device__ __forceinline__ v16h load_b_frag(const _Float16* base, int ld) {
  int lane = threadIdx.x & 31;
  int half = lane >> 4, n = lane & 15;
  FragH f;
#pragma unroll
  for (int i = 0; i < 8; ++i) {
    int k0 = ((i >> 2) << 4) + (half << 3) + ((i & 3) << 1);
    f.u[i] = *(const unsigned*)(base + n * ld + k0);
  }
  return f.v;
}

// B fragment from a pair-packed tile: base[jpair*ld + n] is the dword
// (elem(2*jpair, n), elem(2*jpair+1, n)). Pair i needs k0 = (i>=4?16:0)+half*8+2(i&3),
// i.e. jpair = (i>=4?8:0)+half*4+(i&3) -> one ds_load_b32 per pair, no u16 packing.
__device__ __forceinline__ v16h load_b_frag_pair(const unsigned* base, int ld,
                                                 int ncol) {
  int lane = threadIdx.x & 31;
  int half = lane >> 4, n = lane & 15;
  FragH f;
#pragma unroll
  for (int i = 0; i < 8; ++i) {
    int jp = ((i >> 2) << 3) + (half << 2) + (i & 3);
    f.u[i] = base[jp * ld + ncol + n];
  }
  return f.v;
}

// ---------------------------------------------------------------------------
// Generic WMMA GEMM: C[M,N] = A[M,256] * B[N,256]^T + bias, K fixed = 256.
// Block tile 64x128, 8 waves of 32x32 (2x2 WMMA tiles). LDS strides padded to
// 40 f16 (80B = 5*16B: aligned b128 staging, conflict-free dword frag reads).
// EPI: 0 = store f16, 1 = store f32, 2 = relu -> f16, 3 = relu + residual -> f32
// ---------------------------------------------------------------------------
template <int EPI>
__global__ __launch_bounds__(256) void gemm_wmma(
    const _Float16* __restrict__ A, const _Float16* __restrict__ B,
    const float* __restrict__ bias, const float* __restrict__ res,
    float* __restrict__ outF, _Float16* __restrict__ outH, int M, int N) {
  constexpr int K = 256, BK = 32, LDT = 40;
  __shared__ _Float16 At[64 * LDT];
  __shared__ _Float16 Bt[128 * LDT];

  const int tid = threadIdx.x;
  const int lane = tid & 31, wave = tid >> 5;
  const int wm = wave >> 2, wn = wave & 3;             // 2 x 4 wave grid
  const int mbase = blockIdx.y * 64, nbase = blockIdx.x * 128;

  v8f acc[2][2] = {};

  for (int kb = 0; kb < K; kb += BK) {
    {  // stage A block 64x32 (one 16B copy per thread)
      int row = tid >> 2, cg = tid & 3;
      const _Float16* src = A + (size_t)(mbase + row) * K + kb + cg * 8;
      stage16(src, At + row * LDT + cg * 8);
    }
    {  // stage B block 128x32 (two 16B copies per thread)
      int row = tid >> 1, hc = tid & 1;
      const _Float16* src = B + (size_t)(nbase + row) * K + kb + hc * 16;
      stage16(src,     Bt + row * LDT + hc * 16);
      stage16(src + 8, Bt + row * LDT + hc * 16 + 8);
    }
    stage_wait();
    __syncthreads();

    v16h a0 = load_a_frag(At + (wm * 32) * LDT, LDT);
    v16h a1 = load_a_frag(At + (wm * 32 + 16) * LDT, LDT);
    v16h b0 = load_b_frag(Bt + (wn * 32) * LDT, LDT);
    v16h b1 = load_b_frag(Bt + (wn * 32 + 16) * LDT, LDT);
    acc[0][0] = wmma_f32_f16(a0, b0, acc[0][0]);
    acc[0][1] = wmma_f32_f16(a0, b1, acc[0][1]);
    acc[1][0] = wmma_f32_f16(a1, b0, acc[1][0]);
    acc[1][1] = wmma_f32_f16(a1, b1, acc[1][1]);
    __syncthreads();
  }

  const int half = lane >> 4, nc = lane & 15;
#pragma unroll
  for (int ni = 0; ni < 2; ++ni) {
    int col = nbase + wn * 32 + ni * 16 + nc;
    float bv = bias[col];
#pragma unroll
    for (int mi = 0; mi < 2; ++mi) {
#pragma unroll
      for (int r = 0; r < 8; ++r) {                    // C layout: M = r + 8*half
        int rowg = mbase + wm * 32 + mi * 16 + r + half * 8;
        size_t off = (size_t)rowg * N + col;
        float v = acc[mi][ni][r] + bv;
        if (EPI == 2 || EPI == 3) v = fmaxf(v, 0.f);
        if (EPI == 3) v += res[off];
        if (EPI == 0 || EPI == 2) outH[off] = (_Float16)v;
        else                      outF[off] = v;
      }
    }
  }
}

// ---------------------------------------------------------------------------
// Flash attention over edges. qkv16: [E, 768] f16 (q|k|v, head-major 4x64).
// Block = (q-block of 128, head). Each wave owns 16 query rows; Q fragments
// stay in registers. Stream 32-key K/V tiles through LDS; online softmax with
// 16-lane shfl reductions (C rows split lanes 0-15 / 16-31); P goes through a
// per-wave LDS buffer (same-wave DS ops are in-order) into A-fragment layout.
// V is staged pair-packed so P@V B-fragments are pure ds_load_b32.
// ---------------------------------------------------------------------------
__global__ __launch_bounds__(256) void flash_attn(
    const _Float16* __restrict__ qkv, _Float16* __restrict__ ctx) {
  constexpr int LK = 72;   // K rows: 144B = 9*16B aligned, odd-bank stride
  constexpr int LV = 68;   // V pair rows, in dwords: 272B = 17*16B
  constexpr int LP = 36;
  __shared__ _Float16 Kt[32 * LK];
  __shared__ unsigned Vp[16 * LV];
  __shared__ _Float16 Pt[8 * 16 * LP];

  const int tid = threadIdx.x;
  const int lane = tid & 31, wave = tid >> 5;
  const int half = lane >> 4, l15 = lane & 15;
  const int h = blockIdx.y;
  const int qbase = blockIdx.x * 128 + wave * 16;

  // Q fragments for d=0..31 and d=32..63 (dword pair loads, reused all blocks)
  FragH q0, q1;
#pragma unroll
  for (int i = 0; i < 8; ++i) {
    int k0 = ((i >> 2) << 4) + (half << 3) + ((i & 3) << 1);
    size_t ro = (size_t)(qbase + l15) * (3 * kH) + h * kHD;
    q0.u[i] = *(const unsigned*)(qkv + ro + k0);
    q1.u[i] = *(const unsigned*)(qkv + ro + 32 + k0);
  }

  float mrow[8], lrow[8];
#pragma unroll
  for (int r = 0; r < 8; ++r) { mrow[r] = -1e30f; lrow[r] = 0.f; }
  v8f o[4] = {};

  for (int kb = 0; kb < kEdges; kb += 32) {
    __syncthreads();  // all waves done reading previous K/V tiles
    {  // K tile [32 keys][64 d], one 16B copy per thread
      int row = tid >> 3, cg = tid & 7;
      size_t base = (size_t)(kb + row) * (3 * kH) + kH + h * kHD + cg * 8;
      stage16(qkv + base, Kt + row * LK + cg * 8);
    }
    {  // V tile pair-packed: Vp[jp][d] = (V[2jp][d], V[2jp+1][d]) as one dword
      int jp = tid >> 4, dq = tid & 15;   // 16 pairs x 16 d-quads
      size_t b0 = (size_t)(kb + 2 * jp) * (3 * kH) + 2 * kH + h * kHD + dq * 4;
      v4h a = *(const v4h*)(qkv + b0);
      v4h b = *(const v4h*)(qkv + b0 + 3 * kH);
      v4u p;
#pragma unroll
      for (int w = 0; w < 4; ++w) {
        union { _Float16 hh[2]; unsigned u; } pk;
        pk.hh[0] = a[w]; pk.hh[1] = b[w];
        p[w] = pk.u;
      }
      *(v4u*)(Vp + jp * LV + dq * 4) = p;
    }
    stage_wait();
    __syncthreads();

    // S = Q K^T for 32 keys: two 16-key column tiles x two d-halves
    v8f s0 = {}, s1 = {};
    s0 = wmma_f32_f16(q0.v, load_b_frag(Kt, LK), s0);
    s0 = wmma_f32_f16(q1.v, load_b_frag(Kt + 32, LK), s0);
    s1 = wmma_f32_f16(q0.v, load_b_frag(Kt + 16 * LK, LK), s1);
    s1 = wmma_f32_f16(q1.v, load_b_frag(Kt + 16 * LK + 32, LK), s1);

    float alpha[8];
#pragma unroll
    for (int r = 0; r < 8; ++r) {
      float a = s0[r] * 0.125f, b = s1[r] * 0.125f;   // 1/sqrt(64)
      float mx = fmaxf(a, b);
      mx = fmaxf(mx, __shfl_xor(mx, 1, 32));
      mx = fmaxf(mx, __shfl_xor(mx, 2, 32));
      mx = fmaxf(mx, __shfl_xor(mx, 4, 32));
      mx = fmaxf(mx, __shfl_xor(mx, 8, 32));
      float mn = fmaxf(mrow[r], mx);
      float p0 = __expf(a - mn), p1 = __expf(b - mn);
      s0[r] = p0; s1[r] = p1;
      float sm = p0 + p1;
      sm += __shfl_xor(sm, 1, 32);
      sm += __shfl_xor(sm, 2, 32);
      sm += __shfl_xor(sm, 4, 32);
      sm += __shfl_xor(sm, 8, 32);
      float al = __expf(mrow[r] - mn);
      alpha[r] = al;
      lrow[r] = lrow[r] * al + sm;
      mrow[r] = mn;
    }
#pragma unroll
    for (int t = 0; t < 4; ++t)
#pragma unroll
      for (int r = 0; r < 8; ++r) o[t][r] *= alpha[r];

    // C-layout P -> per-wave LDS -> A-fragment layout (no barrier: same wave)
    _Float16* P = Pt + wave * 16 * LP;
#pragma unroll
    for (int r = 0; r < 8; ++r) {
      P[(r + half * 8) * LP + l15]      = (_Float16)s0[r];
      P[(r + half * 8) * LP + 16 + l15] = (_Float16)s1[r];
    }
    v16h pf = load_a_frag(P, LP);
    o[0] = wmma_f32_f16(pf, load_b_frag_pair(Vp, LV, 0),  o[0]);
    o[1] = wmma_f32_f16(pf, load_b_frag_pair(Vp, LV, 16), o[1]);
    o[2] = wmma_f32_f16(pf, load_b_frag_pair(Vp, LV, 32), o[2]);
    o[3] = wmma_f32_f16(pf, load_b_frag_pair(Vp, LV, 48), o[3]);
  }

#pragma unroll
  for (int t = 0; t < 4; ++t)
#pragma unroll
    for (int r = 0; r < 8; ++r) {
      int row = qbase + r + half * 8;
      ctx[(size_t)row * kH + h * kHD + t * 16 + l15] =
          (_Float16)(o[t][r] / lrow[r]);
    }
}

// out = LayerNorm(x + y) * g + b; one wave per 256-wide row. Optional f16 copy.
__global__ __launch_bounds__(256) void add_layernorm(
    const float* __restrict__ x, const float* __restrict__ y,
    const float* __restrict__ g, const float* __restrict__ b,
    float* __restrict__ outF, _Float16* __restrict__ outH) {
  const int lane = threadIdx.x & 31, wave = threadIdx.x >> 5;
  const size_t row = (size_t)blockIdx.x * 8 + wave;
  float v[8], s = 0.f;
#pragma unroll
  for (int j = 0; j < 8; ++j) {
    int c = j * 32 + lane;
    v[j] = x[row * kH + c] + y[row * kH + c];
    s += v[j];
  }
  for (int m = 1; m < 32; m <<= 1) s += __shfl_xor(s, m, 32);
  float mu = s * (1.f / kH);
  float var = 0.f;
#pragma unroll
  for (int j = 0; j < 8; ++j) { float d = v[j] - mu; var += d * d; }
  for (int m = 1; m < 32; m <<= 1) var += __shfl_xor(var, m, 32);
  float rs = rsqrtf(var * (1.f / kH) + 1e-5f);
#pragma unroll
  for (int j = 0; j < 8; ++j) {
    int c = j * 32 + lane;
    float o = (v[j] - mu) * rs * g[c] + b[c];
    outF[row * kH + c] = o;
    if (outH) outH[row * kH + c] = (_Float16)o;
  }
}

__global__ void cvt_f32_f16(const float* __restrict__ s, _Float16* __restrict__ d,
                            int n) {
  int i = blockIdx.x * 256 + threadIdx.x;
  if (i < n) d[i] = (_Float16)s[i];
}

__global__ void scatter_add(const float* __restrict__ e, const int* __restrict__ dst,
                            float* __restrict__ seg, float* __restrict__ cnt) {
  int idx = blockIdx.x * 256 + threadIdx.x;   // over E*H
  int ed = idx >> 8, c = idx & 255;
  int d = dst[ed];
  atomicAdd(&seg[(size_t)d * kH + c], e[idx]);
  if (c == 0) atomicAdd(&cnt[d], 1.f);
}

__global__ void seg_mean_f16(const float* __restrict__ seg,
                             const float* __restrict__ cnt,
                             _Float16* __restrict__ u) {
  int idx = blockIdx.x * 256 + threadIdx.x;   // over N*H
  int n = idx >> 8;
  u[idx] = (_Float16)(seg[idx] / fmaxf(cnt[n], 1.f));
}

}  // namespace

extern "C" void kernel_launch(void* const* d_in, const int* in_sizes, int n_in,
                              void* d_out, int out_size, void* d_ws, size_t ws_size,
                              hipStream_t stream) {
  (void)in_sizes; (void)n_in; (void)out_size; (void)ws_size;

  const float* node_f = (const float*)d_in[0];
  const float* edge_f = (const float*)d_in[1];
  const float* w_in   = (const float*)d_in[2];
  const float* b_in   = (const float*)d_in[3];
  const float* w_out  = (const float*)d_in[4];
  const float* b_out  = (const float*)d_in[5];
  const float* w1     = (const float*)d_in[6];
  const float* b1     = (const float*)d_in[7];
  const float* w2     = (const float*)d_in[8];
  const float* b2     = (const float*)d_in[9];
  const float* w_nu   = (const float*)d_in[10];
  const float* b_nu   = (const float*)d_in[11];
  const float* ln1g   = (const float*)d_in[12];
  const float* ln1b   = (const float*)d_in[13];
  const float* ln2g   = (const float*)d_in[14];
  const float* ln2b   = (const float*)d_in[15];
  const int*   dst    = (const int*)d_in[16] + kEdges;  // edge_index[1]

  float* n_out = (float*)d_out;                   // [2048, 256]
  float* e_out = n_out + (size_t)kNodes * kH;     // [8192, 256]

  char* ws = (char*)d_ws;
  size_t off = 0;
  auto alloc = [&](size_t bytes) {
    void* p = ws + off;
    off = (off + bytes + 255) & ~(size_t)255;
    return p;
  };
  _Float16* x16   = (_Float16*)alloc((size_t)kEdges * kH * 2);
  _Float16* win16 = (_Float16*)alloc((size_t)3 * kH * kH * 2);
  _Float16* wo16  = (_Float16*)alloc((size_t)kH * kH * 2);
  _Float16* w116  = (_Float16*)alloc((size_t)kH * kH * 2);
  _Float16* w216  = (_Float16*)alloc((size_t)kH * kH * 2);
  _Float16* wnu16 = (_Float16*)alloc((size_t)kH * kH * 2);
  _Float16* qkv16 = (_Float16*)alloc((size_t)kEdges * 3 * kH * 2);
  _Float16* ctx16 = (_Float16*)alloc((size_t)kEdges * kH * 2);
  float*    att   = (float*)   alloc((size_t)kEdges * kH * 4);
  float*    e1f   = (float*)   alloc((size_t)kEdges * kH * 4);
  _Float16* e1h   = (_Float16*)alloc((size_t)kEdges * kH * 2);
  _Float16* hid16 = (_Float16*)alloc((size_t)kEdges * kH * 2);
  float*    mlpo  = (float*)   alloc((size_t)kEdges * kH * 4);
  float*    seg   = (float*)   alloc(((size_t)kNodes * kH + kNodes) * 4);
  float*    cnt   = seg + (size_t)kNodes * kH;
  _Float16* u16   = (_Float16*)alloc((size_t)kNodes * kH * 2);

  // f32 -> f16 conversions (activations + weights)
  cvt_f32_f16<<<(kEdges * kH + 255) / 256, 256, 0, stream>>>(edge_f, x16, kEdges * kH);
  cvt_f32_f16<<<(3 * kH * kH + 255) / 256, 256, 0, stream>>>(w_in, win16, 3 * kH * kH);
  cvt_f32_f16<<<(kH * kH + 255) / 256, 256, 0, stream>>>(w_out, wo16, kH * kH);
  cvt_f32_f16<<<(kH * kH + 255) / 256, 256, 0, stream>>>(w1, w116, kH * kH);
  cvt_f32_f16<<<(kH * kH + 255) / 256, 256, 0, stream>>>(w2, w216, kH * kH);
  cvt_f32_f16<<<(kH * kH + 255) / 256, 256, 0, stream>>>(w_nu, wnu16, kH * kH);

  // qkv = x @ in_proj_w^T + b           [8192, 768] f16
  gemm_wmma<0><<<dim3(3 * kH / 128, kEdges / 64), 256, 0, stream>>>(
      x16, win16, b_in, nullptr, nullptr, qkv16, kEdges, 3 * kH);

  // flash attention -> ctx              [8192, 256] f16
  flash_attn<<<dim3(kEdges / 128, kNH), 256, 0, stream>>>(qkv16, ctx16);

  // attended = ctx @ out_proj_w^T + b   [8192, 256] f32
  gemm_wmma<1><<<dim3(kH / 128, kEdges / 64), 256, 0, stream>>>(
      ctx16, wo16, b_out, nullptr, att, nullptr, kEdges, kH);

  // e1 = LN(edge + attended)
  add_layernorm<<<kEdges / 8, 256, 0, stream>>>(edge_f, att, ln1g, ln1b, e1f, e1h);

  // mlp = relu(e1 @ w1^T + b1) @ w2^T + b2
  gemm_wmma<2><<<dim3(kH / 128, kEdges / 64), 256, 0, stream>>>(
      e1h, w116, b1, nullptr, nullptr, hid16, kEdges, kH);
  gemm_wmma<1><<<dim3(kH / 128, kEdges / 64), 256, 0, stream>>>(
      hid16, w216, b2, nullptr, mlpo, nullptr, kEdges, kH);

  // e2 = LN(e1 + mlp) -> final edge output
  add_layernorm<<<kEdges / 8, 256, 0, stream>>>(e1f, mlpo, ln2g, ln2b, e_out, nullptr);

  // scatter-mean over dst
  (void)hipMemsetAsync(seg, 0, ((size_t)kNodes * kH + kNodes) * 4, stream);
  scatter_add<<<kEdges * kH / 256, 256, 0, stream>>>(e_out, dst, seg, cnt);
  seg_mean_f16<<<kNodes * kH / 256, 256, 0, stream>>>(seg, cnt, u16);

  // n = node_features + relu(u @ nu_w^T + b)
  gemm_wmma<3><<<dim3(kH / 128, kNodes / 64), 256, 0, stream>>>(
      u16, wnu16, b_nu, node_f, n_out, nullptr, kNodes, kH);
}
